// PhiMoESparseMoeBlock_62079457296769
// MI455X (gfx1250) — matrix operand order repaired and checked by reading, not compiled, not run on hardware
//
#include <hip/hip_runtime.h>
#include <hip/hip_bf16.h>
#include <math.h>

typedef __bf16 bf16_t;
typedef __attribute__((ext_vector_type(16))) __bf16 v16bf;
typedef __attribute__((ext_vector_type(8)))  float  v8f;

#define TT 2048   // tokens
#define HH 1024   // hidden
#define FF 4096   // ffn
#define EE 8      // experts
#define ROWTILE 64            // rows per GEMM block (4 WMMA sub-tiles)
#define MAXROWS (2*TT + EE*ROWTILE)   // 4608: 64-aligned per-expert buckets

// ---------------------------------------------------------------------------
// 1) Router: logits = x @ gate_w; sparse-mixer top-2 multipliers; bucket slots
// ---------------------------------------------------------------------------
__global__ void router_kernel(const float* __restrict__ x,
                              const float* __restrict__ gw,
                              int* __restrict__ cnt,
                              int* __restrict__ sel,
                              int* __restrict__ slot,
                              float* __restrict__ mult,
                              int* __restrict__ tokbucket)
{
  const int wavesPerBlk = blockDim.x >> 5;
  const int t    = blockIdx.x * wavesPerBlk + (threadIdx.x >> 5);
  const int lane = threadIdx.x & 31;
  if (t >= TT) return;

  float acc[EE];
#pragma unroll
  for (int e = 0; e < EE; ++e) acc[e] = 0.f;

  const float* xr = x + (size_t)t * HH;
  for (int h = lane; h < HH; h += 32) {
    float xv = xr[h];
#pragma unroll
    for (int e = 0; e < EE; ++e) acc[e] += xv * gw[h * EE + e];
  }
#pragma unroll
  for (int e = 0; e < EE; ++e) {
#pragma unroll
    for (int off = 16; off > 0; off >>= 1)
      acc[e] += __shfl_xor(acc[e], off, 32);
  }

  if (lane == 0) {
    // top-1 / top-2 (first index wins ties, matching lax.top_k)
    int i1 = 0; float v1 = acc[0];
#pragma unroll
    for (int e = 1; e < EE; ++e) if (acc[e] > v1) { v1 = acc[e]; i1 = e; }
    int i2 = 0; float v2 = -INFINITY;
#pragma unroll
    for (int e = 0; e < EE; ++e) if (e != i1 && acc[e] > v2) { v2 = acc[e]; i2 = e; }

    // masked softmax multipliers (JITTER = 0.01 -> threshold 0.02)
    float sum1 = 0.f, sum2 = 0.f;
#pragma unroll
    for (int e = 0; e < EE; ++e) {
      float le = acc[e];
      float f1 = fmaxf(fabsf(le), v1);
      if (!((v1 - le) / f1 > 0.02f)) sum1 += __expf(le - v1);
      if (e != i1) {
        float f2 = fmaxf(fabsf(le), v2);
        if (!((v2 - le) / f2 > 0.02f)) sum2 += __expf(le - v2);
      }
    }
    float m1 = 1.f / sum1;   // exp(v1-v1)/sum1
    float m2 = 1.f / sum2;   // exp(v2-v2)/sum2

    int s1 = atomicAdd(&cnt[i1], 1);
    int s2 = atomicAdd(&cnt[i2], 1);
    sel [t*2+0] = i1; sel [t*2+1] = i2;
    slot[t*2+0] = s1; slot[t*2+1] = s2;
    mult[t*2+0] = m1; mult[t*2+1] = m2;
    tokbucket[i1 * TT + s1] = t;
    tokbucket[i2 * TT + s2] = t;
  }
}

// ---------------------------------------------------------------------------
// 2) ROWTILE-aligned exclusive prefix of counts
// ---------------------------------------------------------------------------
__global__ void prefix_kernel(const int* __restrict__ cnt, int* __restrict__ base)
{
  if (threadIdx.x == 0) {
    int b = 0;
#pragma unroll
    for (int e = 0; e < EE; ++e) {
      base[e] = b;
      b += (cnt[e] + (ROWTILE - 1)) & ~(ROWTILE - 1);
    }
    base[EE] = b;
  }
}

// ---------------------------------------------------------------------------
// 3) Gather token rows -> bf16, per-expert bucketed; zero pad rows
// ---------------------------------------------------------------------------
__global__ void gather_kernel(const float* __restrict__ x,
                              const int* __restrict__ cnt,
                              const int* __restrict__ base,
                              const int* __restrict__ tokbucket,
                              bf16_t* __restrict__ xg)
{
  const int row = blockIdx.x;
  if (row >= base[EE]) return;
  int e = 0;
  while (e < EE - 1 && row >= base[e + 1]) ++e;
  const int s = row - base[e];
  bf16_t* dst = xg + (size_t)row * HH;
  if (s < cnt[e]) {
    const int t = tokbucket[e * TT + s];
    const float* src = x + (size_t)t * HH;
    for (int h = threadIdx.x; h < HH; h += blockDim.x) dst[h] = (bf16_t)src[h];
  } else {
    for (int h = threadIdx.x; h < HH; h += blockDim.x) dst[h] = (bf16_t)0.f;
  }
}

// ---------------------------------------------------------------------------
// WMMA fragment K-maps (CDNA5 ISA 7.12.2, wave32)
//   A (16x32, 16-bit): lane = half*16 + m ; elem i -> K = ((i&8)<<1)+half*8+(i&7)
//   B (32x16, 16-bit): lane = half*16 + n ; elem i -> K = ((i&8)<<1)+half*8+(i&7)
//   C/D (16x16 f32):   lane = half*16 + n ; vgpr r -> M = r + half*8
// ---------------------------------------------------------------------------

// 4) Up-projection: hbuf = silu(Xg @ W1e) * (Xg @ W3e)
//    64 rows x 64 cols per block; 4 A sub-tiles share each B fragment.
__global__ void moe_up_kernel(const float* __restrict__ w1,
                              const float* __restrict__ w3,
                              const bf16_t* __restrict__ xg,
                              bf16_t* __restrict__ hbuf,
                              const int* __restrict__ cnt,
                              const int* __restrict__ base)
{
  const int e  = blockIdx.y >> 5;
  const int rt = blockIdx.y & 31;
  if (rt * ROWTILE >= cnt[e]) return;     // uniform: EXEC stays all-1s
  const int rowbase = base[e] + rt * ROWTILE;

  const int wave = threadIdx.x >> 5;
  const int lane = threadIdx.x & 31;
  const int half = lane >> 4;
  const int l15  = lane & 15;
  const int col  = blockIdx.x * 64 + wave * 16 + l15;

  const float* w1p = w1 + (size_t)e * HH * FF;
  const float* w3p = w3 + (size_t)e * HH * FF;
  const bf16_t* arow = xg + (size_t)(rowbase + l15) * HH + half * 8;

  v8f acc1[4], acc3[4];
#pragma unroll
  for (int st = 0; st < 4; ++st) { acc1[st] = (v8f){}; acc3[st] = (v8f){}; }

  for (int k0 = 0; k0 < HH; k0 += 32) {
    v16bf b1, b3;
#pragma unroll
    for (int i = 0; i < 16; ++i) {        // B: same K map as A, stride FF
      const size_t K = (size_t)(k0 + ((i & 8) << 1) + half * 8 + (i & 7));
      b1[i] = (bf16_t)w1p[K * FF + col];
      b3[i] = (bf16_t)w3p[K * FF + col];
    }
#pragma unroll
    for (int st = 0; st < 4; ++st) {
      const bf16_t* ap = arow + (size_t)st * 16 * HH + k0;
      v16bf a;
#pragma unroll
      for (int i = 0; i < 8; ++i) {       // two contiguous 16B runs
        a[i]     = ap[i];
        a[8 + i] = ap[16 + i];
      }
      acc1[st] = __builtin_amdgcn_wmma_f32_16x16x32_bf16(false, a, false, b1,
                                                         (short)0, acc1[st], false, false);
      acc3[st] = __builtin_amdgcn_wmma_f32_16x16x32_bf16(false, a, false, b3,
                                                         (short)0, acc3[st], false, false);
    }
  }
#pragma unroll
  for (int st = 0; st < 4; ++st) {
#pragma unroll
    for (int r = 0; r < 8; ++r) {
      const int m = st * 16 + r + half * 8;
      const float u = acc1[st][r];
      // silu via v_rcp_f32 (avoids IEEE div expansion)
      const float h = u * __builtin_amdgcn_rcpf(1.f + __expf(-u)) * acc3[st][r];
      hbuf[(size_t)(rowbase + m) * FF + col] = (bf16_t)h;
    }
  }
}

// 5) Down-projection: ybuf = Hbuf @ W2e   (f32 out), same 64x64 blocking
__global__ void moe_down_kernel(const float* __restrict__ w2,
                                const bf16_t* __restrict__ hbuf,
                                float* __restrict__ ybuf,
                                const int* __restrict__ cnt,
                                const int* __restrict__ base)
{
  const int e  = blockIdx.y >> 5;
  const int rt = blockIdx.y & 31;
  if (rt * ROWTILE >= cnt[e]) return;
  const int rowbase = base[e] + rt * ROWTILE;

  const int wave = threadIdx.x >> 5;
  const int lane = threadIdx.x & 31;
  const int half = lane >> 4;
  const int l15  = lane & 15;
  const int col  = blockIdx.x * 64 + wave * 16 + l15;

  const float* w2p = w2 + (size_t)e * FF * HH;
  const bf16_t* arow = hbuf + (size_t)(rowbase + l15) * FF + half * 8;

  v8f acc[4];
#pragma unroll
  for (int st = 0; st < 4; ++st) acc[st] = (v8f){};

  for (int k0 = 0; k0 < FF; k0 += 32) {
    v16bf b;
#pragma unroll
    for (int i = 0; i < 16; ++i) {
      const size_t K = (size_t)(k0 + ((i & 8) << 1) + half * 8 + (i & 7));
      b[i] = (bf16_t)w2p[K * HH + col];
    }
#pragma unroll
    for (int st = 0; st < 4; ++st) {
      const bf16_t* ap = arow + (size_t)st * 16 * FF + k0;
      v16bf a;
#pragma unroll
      for (int i = 0; i < 8; ++i) {
        a[i]     = ap[i];
        a[8 + i] = ap[16 + i];
      }
      acc[st] = __builtin_amdgcn_wmma_f32_16x16x32_bf16(false, a, false, b,
                                                        (short)0, acc[st], false, false);
    }
  }
#pragma unroll
  for (int st = 0; st < 4; ++st) {
#pragma unroll
    for (int r = 0; r < 8; ++r) {
      const int m = st * 16 + r + half * 8;
      ybuf[(size_t)(rowbase + m) * HH + col] = acc[st][r];
    }
  }
}

// ---------------------------------------------------------------------------
// 6) Combine: out[t] = m0 * y[base[e0]+s0] + m1 * y[base[e1]+s1]
// ---------------------------------------------------------------------------
__global__ void combine_kernel(const float* __restrict__ ybuf,
                               const int* __restrict__ base,
                               const int* __restrict__ sel,
                               const int* __restrict__ slot,
                               const float* __restrict__ mult,
                               float* __restrict__ out)
{
  const int idx = blockIdx.x * blockDim.x + threadIdx.x;
  const int t = idx >> 10;          // / HH
  const int h = idx & (HH - 1);
  const int r0 = base[sel[t*2+0]] + slot[t*2+0];
  const int r1 = base[sel[t*2+1]] + slot[t*2+1];
  out[idx] = mult[t*2+0] * ybuf[(size_t)r0 * HH + h]
           + mult[t*2+1] * ybuf[(size_t)r1 * HH + h];
}

// ---------------------------------------------------------------------------
extern "C" void kernel_launch(void* const* d_in, const int* in_sizes, int n_in,
                              void* d_out, int out_size, void* d_ws, size_t ws_size,
                              hipStream_t stream)
{
  (void)in_sizes; (void)n_in; (void)out_size; (void)ws_size;
  const float* x  = (const float*)d_in[0];   // [1,2048,1024]
  const float* gw = (const float*)d_in[1];   // [1024,8]
  const float* w1 = (const float*)d_in[2];   // [8,1024,4096]
  const float* w2 = (const float*)d_in[3];   // [8,4096,1024]
  const float* w3 = (const float*)d_in[4];   // [8,1024,4096]
  float* out = (float*)d_out;

  char* p = (char*)d_ws;
  auto alloc = [&](size_t bytes) {
    char* r = p;
    p += (bytes + 255) & ~(size_t)255;
    return r;
  };
  int*    cnt       = (int*)   alloc(EE * sizeof(int));
  int*    base      = (int*)   alloc((EE + 1) * sizeof(int));
  int*    sel       = (int*)   alloc((size_t)TT * 2 * sizeof(int));
  int*    slot      = (int*)   alloc((size_t)TT * 2 * sizeof(int));
  float*  mult      = (float*) alloc((size_t)TT * 2 * sizeof(float));
  int*    tokbucket = (int*)   alloc((size_t)EE * TT * sizeof(int));
  bf16_t* xg        = (bf16_t*)alloc((size_t)MAXROWS * HH * sizeof(bf16_t));
  bf16_t* hbuf      = (bf16_t*)alloc((size_t)MAXROWS * FF * sizeof(bf16_t));
  float*  ybuf      = (float*) alloc((size_t)MAXROWS * HH * sizeof(float));

  hipMemsetAsync(cnt, 0, EE * sizeof(int), stream);
  router_kernel <<<TT / 4, 128, 0, stream>>>(x, gw, cnt, sel, slot, mult, tokbucket);
  prefix_kernel <<<1, 32, 0, stream>>>(cnt, base);
  gather_kernel <<<MAXROWS, 256, 0, stream>>>(x, cnt, base, tokbucket, xg);
  // 64-row x 64-col tiles: grid.y = experts * (2048 / 64)
  moe_up_kernel  <<<dim3(FF / 64, EE * 32), 128, 0, stream>>>(w1, w3, xg, hbuf, cnt, base);
  moe_down_kernel<<<dim3(HH / 64, EE * 32), 128, 0, stream>>>(w2, hbuf, ybuf, cnt, base);
  combine_kernel <<<(TT * HH) / 256, 256, 0, stream>>>(ybuf, base, sel, slot, mult, out);
}
